// RelativeMultiHeadedAttention_71837622993540
// MI455X (gfx1250) — matrix operand order repaired and checked by reading, not compiled, not run on hardware
//
#include <hip/hip_runtime.h>
#include <hip/hip_bf16.h>

typedef __bf16 bf16_t;
typedef unsigned short u16;
typedef unsigned int   u32;
typedef unsigned short v8u  __attribute__((ext_vector_type(8)));
typedef unsigned short v16u __attribute__((ext_vector_type(16)));
typedef __bf16 v16bf __attribute__((ext_vector_type(16)));
typedef __bf16 v4bf  __attribute__((ext_vector_type(4)));
typedef float  v8f   __attribute__((ext_vector_type(8)));
typedef u32    v4u32 __attribute__((ext_vector_type(4)));
typedef u32    v8u32 __attribute__((ext_vector_type(8)));

#define DM   1024
#define SEQ  2048
#define NH   16
#define DK   64
#define BATCH 2
#define NEG_BIG (-3.0e38f)

// ---------------- WMMA fragment helpers (gfx1250 wave32 layouts) ----------------

// A-matrix 16x32 bf16: lane&15 = M row; lanes 0-15 hold K {0..7,16..23}, lanes 16-31 hold K {8..15,24..31}
__device__ __forceinline__ v16bf load_frag_a(const bf16_t* base, int ld, int lane) {
  const int half = lane >> 4;
  const int m    = lane & 15;
  const bf16_t* row = base + (size_t)m * ld;
  v8u lo = *(const v8u*)(row + half * 8);
  v8u hi = *(const v8u*)(row + 16 + half * 8);
  v16u u = __builtin_shufflevector(lo, hi, 0,1,2,3,4,5,6,7,8,9,10,11,12,13,14,15);
  return __builtin_bit_cast(v16bf, u);
}

// B-matrix 32x16 bf16 from "N rows x K contiguous" source: lane&15 = N; lanes 0-15 K {0..7,16..23}, lanes 16-31 K {8..15,24..31}
__device__ __forceinline__ v16bf load_frag_b(const bf16_t* base, int ld, int lane) {
  const int kh = (lane >> 4) * 8;
  const int n  = lane & 15;
  const bf16_t* row = base + (size_t)n * ld + kh;
  v8u lo = *(const v8u*)(row);
  v8u hi = *(const v8u*)(row + 16);
  v16u u = __builtin_shufflevector(lo, hi, 0,1,2,3,4,5,6,7,8,9,10,11,12,13,14,15);
  return __builtin_bit_cast(v16bf, u);
}

__device__ __forceinline__ v8f wmma_bf16f32(v16bf a, v16bf b, v8f c) {
  return __builtin_amdgcn_wmma_f32_16x16x32_bf16(false, a, false, b, (short)0, c, false, false);
}

// ---------------- Tensor Data Mover: 2D tile global -> LDS ----------------
// D# per CDNA5 ISA: group0 (4 SGPRs) + group1 (8 SGPRs).
// tile_d0 = contiguous elements per row, tile_d1 = rows, stride = row stride (elements).
// bf16 elements (data_size code 1 = 2 bytes). Issue from ONE wave only (EXEC ignored).
__device__ __forceinline__ void tdm_load_2d(const void* gptr, const void* lds_ptr,
                                            u32 tile_d0, u32 tile_d1, u32 stride_elems) {
  const unsigned long long ga = (unsigned long long)(size_t)gptr;
  const u32 lds = (u32)(size_t)lds_ptr;
  v4u32 g0;
  g0[0] = 1u;                                               // count=1 (valid), user mode
  g0[1] = lds;                                              // lds_addr (bytes)
  g0[2] = (u32)ga;                                          // global_addr[31:0]
  g0[3] = (u32)((ga >> 32) & 0x1FFFFFFull) | (2u << 30);    // global_addr[56:32], type=2
  v8u32 g1;
  g1[0] = 1u << 16;                                         // workgroup_mask=0, data_size=2B
  g1[1] = (tile_d0 & 0xFFFFu) << 16;                        // tensor_dim0[15:0]
  g1[2] = (tile_d0 >> 16) | ((tile_d1 & 0xFFFFu) << 16);    // tensor_dim0 hi | tensor_dim1 lo
  g1[3] = (tile_d1 >> 16) | ((tile_d0 & 0xFFFFu) << 16);    // tensor_dim1 hi | tile_dim0
  g1[4] = tile_d1 & 0xFFFFu;                                // tile_dim1 (tile_dim2=0)
  g1[5] = stride_elems;                                     // tensor_dim0_stride[31:0]
  g1[6] = 0u;                                               // stride hi | tensor_dim1_stride lo
  g1[7] = 0u;
  asm volatile("tensor_load_to_lds %0, %1" :: "s"(g0), "s"(g1) : "memory");
}

__device__ __forceinline__ void tdm_wait() {
  __builtin_amdgcn_s_wait_tensorcnt(0);
}

// ---------------- fp32 -> bf16 convert (x4 vectorized) ----------------
__global__ __launch_bounds__(256) void cvt_f32_bf16(const float* __restrict__ in,
                                                    bf16_t* __restrict__ out, int n4) {
  int i = blockIdx.x * 256 + threadIdx.x;
  if (i < n4) {
    const float4 v = ((const float4*)in)[i];
    v4bf r;
    r[0] = (bf16_t)v.x; r[1] = (bf16_t)v.y; r[2] = (bf16_t)v.z; r[3] = (bf16_t)v.w;
    ((v4bf*)out)[i] = r;
  }
}

// ---------------- Tiled WMMA GEMM: C(MxN) = A(MxK) * W(NxK)^T + bias ----------------
// Block tile 128x128, K-step 32, 256 threads = 8 waves (4x2 wave grid), each wave 32x64.
// A/W tiles staged into LDS by the Tensor Data Mover (wave 0 issues, TENSORcnt + barrier).
template <int BF16OUT>
__global__ __launch_bounds__(256) void gemm_wmma(const bf16_t* __restrict__ A,
                                                 const bf16_t* __restrict__ W,
                                                 const float* __restrict__ bias,
                                                 void* __restrict__ Cout,
                                                 int M, int N, int K) {
  __shared__ alignas(16) bf16_t As[128 * 32];
  __shared__ alignas(16) bf16_t Ws[128 * 32];
  const int tid  = threadIdx.x;
  const int lane = tid & 31;
  const int wave = tid >> 5;      // 0..7
  const int wm   = wave >> 1;     // 0..3 -> 32 rows each
  const int wn   = wave & 1;      // 0..1 -> 64 cols each
  const int bm0  = blockIdx.y * 128;
  const int bn0  = blockIdx.x * 128;

  v8f acc[2][4];
#pragma unroll
  for (int i = 0; i < 2; ++i)
#pragma unroll
    for (int j = 0; j < 4; ++j) acc[i][j] = {};

  for (int k0 = 0; k0 < K; k0 += 32) {
    if (wave == 0) {
      tdm_load_2d(A + (size_t)bm0 * K + k0, As, 32u, 128u, (u32)K);
      tdm_load_2d(W + (size_t)bn0 * K + k0, Ws, 32u, 128u, (u32)K);
      tdm_wait();
    } else {
      // prefetch next K tile of A into GL2 while the TDM runs
      __builtin_prefetch(A + (size_t)(bm0 + wave * 16) * K + k0 + 32, 0, 0);
    }
    __syncthreads();

    v16bf af[2], bfrag[4];
#pragma unroll
    for (int i = 0; i < 2; ++i)
      af[i] = load_frag_a(As + (wm * 32 + i * 16) * 32, 32, lane);
#pragma unroll
    for (int j = 0; j < 4; ++j)
      bfrag[j] = load_frag_b(Ws + (wn * 64 + j * 16) * 32, 32, lane);
#pragma unroll
    for (int i = 0; i < 2; ++i)
#pragma unroll
      for (int j = 0; j < 4; ++j)
        acc[i][j] = wmma_bf16f32(af[i], bfrag[j], acc[i][j]);
    __syncthreads();
  }

  const int half = lane >> 4;
  const int ln   = lane & 15;
#pragma unroll
  for (int i = 0; i < 2; ++i) {
    const int rbase = bm0 + wm * 32 + i * 16 + half * 8;
#pragma unroll
    for (int j = 0; j < 4; ++j) {
      const int col = bn0 + wn * 64 + j * 16 + ln;
      const float bv = bias[col];
#pragma unroll
      for (int r = 0; r < 8; ++r) {
        const float v = acc[i][j][r] + bv;
        const size_t off = (size_t)(rbase + r) * N + col;
        if (BF16OUT) ((bf16_t*)Cout)[off] = (bf16_t)v;
        else         ((float*)Cout)[off]  = v;
      }
    }
  }
}

// ---------------- QEr and skewed Srel ----------------
__global__ __launch_bounds__(256) void qer_kernel(const bf16_t* __restrict__ Qg,
                                                  const float* __restrict__ Er,
                                                  float* __restrict__ qer) {
  const int idx = blockIdx.x * 256 + threadIdx.x;       // bh*SEQ + l
  const int l  = idx & (SEQ - 1);
  const int bh = idx >> 11;
  const int b = bh >> 4, h = bh & 15;
  const bf16_t* q = Qg + (size_t)(b * SEQ + l) * DM + h * DK;
  const float*  e = Er + (size_t)l * DK;                 // MAX_LEN - S == 0
  float s = 0.f;
#pragma unroll
  for (int d = 0; d < DK; ++d) s += (float)q[d] * e[d];
  qer[idx] = s;
}

__global__ __launch_bounds__(256) void srel_kernel(const float* __restrict__ qer,
                                                   float* __restrict__ srel) {
  const int idx = blockIdx.x * 256 + threadIdx.x;       // bh*SEQ + m
  const int m  = idx & (SEQ - 1);
  const int bh = idx >> 11;
  float v = 0.f;
  if (m & 1) v = qer[bh * SEQ + ((SEQ + m - 1) >> 1)] * 0.125f;  // pre-scaled 1/sqrt(64)
  srel[idx] = v;
}

// ---------------- Flash attention with causal mask + srel bias ----------------
// grid: (SEQ/128, BATCH*NH); block 256 = 8 waves, each wave owns 16 query rows.
// K block staged via TDM; V block transposed manually into LDS.
__global__ __launch_bounds__(256) void attn_kernel(const bf16_t* __restrict__ Qg,
                                                   const bf16_t* __restrict__ Kg,
                                                   const bf16_t* __restrict__ Vg,
                                                   const float* __restrict__ srelAll,
                                                   bf16_t* __restrict__ Og) {
  __shared__ alignas(16) bf16_t Ks[64 * 64];      // key block, (m, d) row-major
  __shared__ alignas(16) bf16_t Vt[64 * 64];      // value block transposed, (d, m)
  __shared__ alignas(16) bf16_t Ps[8][16 * 64];   // per-wave probability scratch

  const int tid = threadIdx.x, lane = tid & 31, wave = tid >> 5;
  const int qb = blockIdx.x;       // query block of 128 rows
  const int bh = blockIdx.y;
  const int b = bh >> 4, h = bh & 15;
  const bf16_t* Qb_ = Qg + (size_t)b * SEQ * DM + h * DK;
  const bf16_t* Kb_ = Kg + (size_t)b * SEQ * DM + h * DK;
  const bf16_t* Vb_ = Vg + (size_t)b * SEQ * DM + h * DK;
  bf16_t*       Ob_ = Og + (size_t)b * SEQ * DM + h * DK;
  const float* srel = srelAll + (size_t)bh * SEQ;

  const int rowW = qb * 128 + wave * 16;     // this wave's first query row
  const int half = lane >> 4, ln = lane & 15;

  v16bf aq[2];
  aq[0] = load_frag_a(Qb_ + (size_t)rowW * DM, DM, lane);
  aq[1] = load_frag_a(Qb_ + (size_t)rowW * DM + 32, DM, lane);

  v8f o[4];
  float mi[8], li[8];
#pragma unroll
  for (int dn = 0; dn < 4; ++dn) o[dn] = {};
#pragma unroll
  for (int r = 0; r < 8; ++r) { mi[r] = NEG_BIG; li[r] = 0.f; }

  const int nkb = qb * 2 + 2;                // uniform per block (causal bound)
  for (int kb = 0; kb < nkb; ++kb) {
    // ---- stage K via TDM (64 rows x 64 cols, row stride DM) ----
    if (wave == 0) {
      tdm_load_2d(Kb_ + (size_t)(kb * 64) * DM, Ks, 64u, 64u, (u32)DM);
    }
    // ---- stage V transposed manually (all 256 threads) ----
    {
      const int crow = tid >> 2;               // 0..63
      const int cc0  = (tid & 3) * 16;         // 0,16,32,48
      const bf16_t* gv = Vb_ + (size_t)(kb * 64 + crow) * DM + cc0;
      v8u v0 = *(const v8u*)gv, v1 = *(const v8u*)(gv + 8);
#pragma unroll
      for (int e = 0; e < 8; ++e) {
        Vt[(cc0 + e) * 64 + crow]     = __builtin_bit_cast(bf16_t, (u16)v0[e]);
        Vt[(cc0 + 8 + e) * 64 + crow] = __builtin_bit_cast(bf16_t, (u16)v1[e]);
      }
    }
    if (wave == 0) tdm_wait();
    __syncthreads();

    // ---- scores: S = Q * K^T (16x64 per wave) ----
    v8f s[4];
#pragma unroll
    for (int j = 0; j < 4; ++j) s[j] = {};
#pragma unroll
    for (int kk = 0; kk < 2; ++kk)
#pragma unroll
      for (int j = 0; j < 4; ++j) {
        v16bf kf = load_frag_b(Ks + (j * 16) * 64 + kk * 32, 64, lane);
        s[j] = wmma_bf16f32(aq[kk], kf, s[j]);
      }

    // ---- scale + srel bias + causal mask, then online softmax ----
    float sv[4][8], bm[8];
#pragma unroll
    for (int r = 0; r < 8; ++r) bm[r] = NEG_BIG;
#pragma unroll
    for (int j = 0; j < 4; ++j) {
      const int col = kb * 64 + j * 16 + ln;
      const float sr = srel[col];
#pragma unroll
      for (int r = 0; r < 8; ++r) {
        const int row = rowW + r + half * 8;
        float v = s[j][r] * 0.125f + sr;
        if (col > row) v = NEG_BIG;
        sv[j][r] = v;
        bm[r] = fmaxf(bm[r], v);
      }
    }
#pragma unroll
    for (int r = 0; r < 8; ++r) {
#pragma unroll
      for (int msk = 8; msk >= 1; msk >>= 1)     // rows live across 16-lane groups
        bm[r] = fmaxf(bm[r], __shfl_xor(bm[r], msk, 32));
    }
    float alpha[8];
#pragma unroll
    for (int r = 0; r < 8; ++r) {
      const float nm = fmaxf(mi[r], bm[r]);
      alpha[r] = __expf(mi[r] - nm);
      mi[r] = nm;
    }
    float psum[8];
#pragma unroll
    for (int r = 0; r < 8; ++r) psum[r] = 0.f;
#pragma unroll
    for (int j = 0; j < 4; ++j) {
      const int cl = j * 16 + ln;
#pragma unroll
      for (int r = 0; r < 8; ++r) {
        const float p = (sv[j][r] > -1.0e37f) ? __expf(sv[j][r] - mi[r]) : 0.f;
        psum[r] += p;
        Ps[wave][(r + half * 8) * 64 + cl] = (bf16_t)p;   // transpose via LDS
      }
    }
#pragma unroll
    for (int r = 0; r < 8; ++r) {
#pragma unroll
      for (int msk = 8; msk >= 1; msk >>= 1)
        psum[r] += __shfl_xor(psum[r], msk, 32);
      li[r] = li[r] * alpha[r] + psum[r];
    }
#pragma unroll
    for (int dn = 0; dn < 4; ++dn)
#pragma unroll
      for (int r = 0; r < 8; ++r) o[dn][r] *= alpha[r];

    asm volatile("s_wait_dscnt 0" ::: "memory");  // wave-local P writes before re-read

    // ---- O += P * V ----
#pragma unroll
    for (int kk = 0; kk < 2; ++kk) {
      v16bf pa = load_frag_a(&Ps[wave][kk * 32], 64, lane);
#pragma unroll
      for (int dn = 0; dn < 4; ++dn) {
        v16bf vf = load_frag_b(Vt + (dn * 16) * 64 + kk * 32, 64, lane);
        o[dn] = wmma_bf16f32(pa, vf, o[dn]);
      }
    }
    __syncthreads();
  }

  // ---- epilogue: normalize and store bf16 ----
#pragma unroll
  for (int r = 0; r < 8; ++r) li[r] = 1.f / li[r];
#pragma unroll
  for (int dn = 0; dn < 4; ++dn)
#pragma unroll
    for (int r = 0; r < 8; ++r) {
      const float v = o[dn][r] * li[r];
      Ob_[(size_t)(rowW + r + half * 8) * DM + dn * 16 + ln] = (bf16_t)v;
    }
}

// ---------------- host-side launcher ----------------
extern "C" void kernel_launch(void* const* d_in, const int* in_sizes, int n_in,
                              void* d_out, int out_size, void* d_ws, size_t ws_size,
                              hipStream_t stream) {
  (void)in_sizes; (void)n_in; (void)out_size; (void)ws_size;
  const float* x  = (const float*)d_in[0];
  const float* Wq = (const float*)d_in[1];
  const float* bq = (const float*)d_in[2];
  const float* Wk = (const float*)d_in[3];
  const float* bk = (const float*)d_in[4];
  const float* Wv = (const float*)d_in[5];
  const float* bv = (const float*)d_in[6];
  const float* Wo = (const float*)d_in[7];
  const float* bo = (const float*)d_in[8];
  const float* Er = (const float*)d_in[9];
  float* out = (float*)d_out;

  const int M = BATCH * SEQ;                 // 4096
  const size_t nX = (size_t)M * DM;          // 4,194,304
  const size_t nW = (size_t)DM * DM;         // 1,048,576
  const size_t nBH = (size_t)BATCH * NH * SEQ;

  char* w = (char*)d_ws;
  size_t off = 0;
  auto alloc = [&](size_t bytes) { void* p = w + off; off = (off + bytes + 255) & ~(size_t)255; return p; };
  bf16_t* xb  = (bf16_t*)alloc(nX * 2);
  bf16_t* Wqb = (bf16_t*)alloc(nW * 2);
  bf16_t* Wkb = (bf16_t*)alloc(nW * 2);
  bf16_t* Wvb = (bf16_t*)alloc(nW * 2);
  bf16_t* Wob = (bf16_t*)alloc(nW * 2);
  bf16_t* Qb  = (bf16_t*)alloc(nX * 2);
  bf16_t* Kb  = (bf16_t*)alloc(nX * 2);
  bf16_t* Vb  = (bf16_t*)alloc(nX * 2);
  bf16_t* Ab  = (bf16_t*)alloc(nX * 2);
  float*  qer  = (float*)alloc(nBH * 4);
  float*  srel = (float*)alloc(nBH * 4);

  // 1) converts
  cvt_f32_bf16<<<(int)(nX / 4 / 256), 256, 0, stream>>>(x,  xb,  (int)(nX / 4));
  cvt_f32_bf16<<<(int)(nW / 4 / 256), 256, 0, stream>>>(Wq, Wqb, (int)(nW / 4));
  cvt_f32_bf16<<<(int)(nW / 4 / 256), 256, 0, stream>>>(Wk, Wkb, (int)(nW / 4));
  cvt_f32_bf16<<<(int)(nW / 4 / 256), 256, 0, stream>>>(Wv, Wvb, (int)(nW / 4));
  cvt_f32_bf16<<<(int)(nW / 4 / 256), 256, 0, stream>>>(Wo, Wob, (int)(nW / 4));

  // 2) QKV projections (bf16 out)
  dim3 ggrid(DM / 128, M / 128);
  gemm_wmma<1><<<ggrid, 256, 0, stream>>>(xb, Wqb, bq, (void*)Qb, M, DM, DM);
  gemm_wmma<1><<<ggrid, 256, 0, stream>>>(xb, Wkb, bk, (void*)Kb, M, DM, DM);
  gemm_wmma<1><<<ggrid, 256, 0, stream>>>(xb, Wvb, bv, (void*)Vb, M, DM, DM);

  // 3) relative-position bias
  qer_kernel<<<(int)(nBH / 256), 256, 0, stream>>>(Qb, Er, qer);
  srel_kernel<<<(int)(nBH / 256), 256, 0, stream>>>(qer, srel);

  // 4) flash attention
  attn_kernel<<<dim3(SEQ / 128, BATCH * NH), 256, 0, stream>>>(Qb, Kb, Vb, srel, Ab);

  // 5) output projection (fp32 out)
  gemm_wmma<0><<<ggrid, 256, 0, stream>>>(Ab, Wob, bo, (void*)out, M, DM, DM);
}
